// SparseBottleneck_89309549953683
// MI455X (gfx1250) — matrix operand order repaired and checked by reading, not compiled, not run on hardware
//
#include <hip/hip_runtime.h>
#include <hip/hip_bf16.h>
#include <stdint.h>

// ---------------------------------------------------------------------------
// SparseBottleneck on gfx1250: bf16 WMMA gather-GEMM-scatter with async-LDS
// staging (GLOBAL_LOAD_ASYNC_TO_LDS_B128 / ASYNCcnt) and double-buffered tiles.
//   c1  = sparse_conv(feats, W1)            (f32 atomic accumulate, L2-resident)
//   c2  = bf16( lrelu( lrelu(c1) @ W2 ) )   (dense WMMA, fused lrelu both sides)
//   out = sparse_conv(c2, W3) * mask        (accumulate into d_out, then mask)
// ---------------------------------------------------------------------------

typedef __attribute__((ext_vector_type(16))) __bf16 v16bf;
typedef __attribute__((ext_vector_type(8)))  __bf16 v8bf;
typedef __attribute__((ext_vector_type(4)))  __bf16 v4bf;
typedef __attribute__((ext_vector_type(8)))  float  v8f;

#define N_PTS    200000
#define M_PAIRS  100000
#define KTAPS    27
#define CCH      128
#define NEG_SLOPE 0.2f
#define PITCH    136          // LDS row pitch in bf16 elems (128 + 8 pad, 272B)
#define MTPB     5            // M-tiles (of 16 rows) per block

__device__ __forceinline__ float lrelu(float x) {
    return x >= 0.f ? x : NEG_SLOPE * x;
}

// --- CDNA5 async global->LDS copy (ASYNCcnt), guarded by __has_builtin ------
#if __has_builtin(__builtin_amdgcn_global_load_async_to_lds_b128)
#define HAVE_ASYNC_LDS 1
typedef int v4i __attribute__((vector_size(16)));
typedef __attribute__((address_space(1))) v4i* gv4i_p;   // global int4*
typedef __attribute__((address_space(3))) v4i* lv4i_p;   // LDS int4*
__device__ __forceinline__ void async_copy16(const void* g, void* l) {
    __builtin_amdgcn_global_load_async_to_lds_b128(
        (gv4i_p)(uintptr_t)g, (lv4i_p)(unsigned)(uintptr_t)l, 0, 0);
}
__device__ __forceinline__ void wait_async0() {
#if __has_builtin(__builtin_amdgcn_s_wait_asynccnt)
    __builtin_amdgcn_s_wait_asynccnt(0);
#else
    asm volatile("s_wait_asynccnt 0x0" ::: "memory");
#endif
}
__device__ __forceinline__ void wait_async1() {
#if __has_builtin(__builtin_amdgcn_s_wait_asynccnt)
    __builtin_amdgcn_s_wait_asynccnt(1);
#else
    asm volatile("s_wait_asynccnt 0x1" ::: "memory");
#endif
}
#else
#define HAVE_ASYNC_LDS 0
#endif

// --- prep kernels -----------------------------------------------------------

__global__ void k_convert_feats(const float* __restrict__ in,
                                __bf16* __restrict__ outb, int total4) {
    int i = blockIdx.x * blockDim.x + threadIdx.x;
    if (i >= total4) return;
    float4 v = ((const float4*)in)[i];
    v4bf o;
    o[0] = (__bf16)v.x; o[1] = (__bf16)v.y; o[2] = (__bf16)v.z; o[3] = (__bf16)v.w;
    ((v4bf*)outb)[i] = o;
}

// Wt[k][n][kk] = bf16(W[k][kk][n])   (N-major so B fragments are contiguous)
__global__ void k_transpose_w(const float* __restrict__ W,
                              __bf16* __restrict__ Wt, int total) {
    int idx = blockIdx.x * blockDim.x + threadIdx.x;
    if (idx >= total) return;
    int k   = idx >> 14;
    int rem = idx & 16383;
    int n   = rem >> 7;
    int kk  = rem & 127;
    Wt[idx] = (__bf16)W[(k << 14) + (kk << 7) + n];
}

__global__ void k_mask(float* __restrict__ out, const float* __restrict__ mask,
                       int total4) {
    int i = blockIdx.x * blockDim.x + threadIdx.x;
    if (i >= total4) return;
    float m  = mask[i >> 5];          // (i*4) >> 7
    float4 v = ((float4*)out)[i];
    v.x *= m; v.y *= m; v.z *= m; v.w *= m;
    ((float4*)out)[i] = v;
}

// --- gather-GEMM-scatter sparse conv ---------------------------------------
// grid = (M_PAIRS/(16*MTPB), KTAPS), block = 256 (8 waves)
__global__ __launch_bounds__(256) void k_sparse_conv(
    const __bf16* __restrict__ X,       // [N, 128] bf16
    const __bf16* __restrict__ Wt,      // [K][128(n)][128(kk)] bf16 (transposed)
    const int*    __restrict__ in_map,  // [K, M]
    const int*    __restrict__ out_map, // [K, M]
    float*        __restrict__ Out)     // [N, 128] f32 accumulator
{
    __shared__ __align__(16) __bf16 Bs[CCH * PITCH];       // 34.8 KB
    __shared__ __align__(16) __bf16 As[2][16 * PITCH];     //  8.7 KB (dbl buf)
    __shared__ int outr[2][16];

    const int t    = threadIdx.x;
    const int lane = t & 31;
    const int lhi  = lane >> 4;        // half-wave select
    const int lm   = lane & 15;
    const int wv   = t >> 5;           // wave id = output column tile
    const int ktap = blockIdx.y;
    const __bf16* Wk = Wt + (size_t)ktap * CCH * CCH;
    const int map_base = ktap * M_PAIRS + blockIdx.x * (MTPB * 16);

    const int sr = t >> 4;             // staging row   0..15
    const int sc = t & 15;             // staging chunk 0..15 (16B each)

    // ---- stage full 128x128 bf16 weight tile (n-major): 2048 x 16B chunks
    #pragma unroll
    for (int p = 0; p < 8; ++p) {
        int chk = p * 256 + t;
        __bf16*       dst = &Bs[(chk >> 4) * PITCH + (chk & 15) * 8];
        const __bf16* src = Wk + chk * 8;
#if HAVE_ASYNC_LDS
        async_copy16(src, dst);
#else
        *(uint4*)dst = *(const uint4*)src;
#endif
    }

    // ---- stage gathered A tile 0 (16 rows x 256B)
    {
        int row = in_map[map_base + sr];
        const __bf16* src = X + (size_t)row * CCH + sc * 8;
        __bf16*       dst = &As[0][sr * PITCH + sc * 8];
#if HAVE_ASYNC_LDS
        async_copy16(src, dst);
#else
        *(uint4*)dst = *(const uint4*)src;
#endif
        if (t < 16) outr[0][t] = out_map[map_base + t];
    }

#if HAVE_ASYNC_LDS
    wait_async0();
#endif
    __syncthreads();

    // ---- preload B fragments into VGPRs, reused across all M-tiles.
    // B layout used here (16-bit B 32x16): lanes 0-15: N=lane, K=0..15 contiguous;
    // lanes 16-31: K=16..31.
    const int ncol = wv * 16 + lm;
    v16bf bfrag[4];
    #pragma unroll
    for (int s = 0; s < 4; ++s) {
        const __bf16* bp = &Bs[ncol * PITCH + s * 32 + lhi * 16];
        v8bf lo = *(const v8bf*)(bp);
        v8bf hi = *(const v8bf*)(bp + 8);
        #pragma unroll
        for (int j = 0; j < 8; ++j) { bfrag[s][j] = lo[j]; bfrag[s][j + 8] = hi[j]; }
    }

    for (int it = 0; it < MTPB; ++it) {
        const int cur = it & 1;

        // issue async gather for the next tile into the other buffer
        if (it + 1 < MTPB) {
            const int nb = map_base + (it + 1) * 16;
            int row = in_map[nb + sr];
            const __bf16* src = X + (size_t)row * CCH + sc * 8;
            __bf16*       dst = &As[cur ^ 1][sr * PITCH + sc * 8];
#if HAVE_ASYNC_LDS
            async_copy16(src, dst);
#else
            *(uint4*)dst = *(const uint4*)src;
#endif
            if (t < 16) outr[cur ^ 1][t] = out_map[nb + t];
#if HAVE_ASYNC_LDS
            wait_async1();    // tile `it` landed (in-order); next may be in flight
#endif
        }
#if HAVE_ASYNC_LDS
        else wait_async0();
#endif
        __syncthreads();      // tile `it` visible to all waves

        // 16x128 @ 128x16 via 4 chained wmma (K=32 each)
        // A layout (16-bit A 16x32, matches B): lanes 0-15: M=lane, K=0..15 contiguous;
        // lanes 16-31: K=16..31.
        v8f acc = {};
        #pragma unroll
        for (int s = 0; s < 4; ++s) {
            const __bf16* ap = &As[cur][lm * PITCH + s * 32 + lhi * 16];
            v8bf lo = *(const v8bf*)(ap);
            v8bf hi = *(const v8bf*)(ap + 8);
            v16bf afrag;
            #pragma unroll
            for (int j = 0; j < 8; ++j) { afrag[j] = lo[j]; afrag[j + 8] = hi[j]; }
            acc = __builtin_amdgcn_wmma_f32_16x16x32_bf16(
                      false, afrag, false, bfrag[s], (short)0, acc, false, false);
        }

        // scatter: C/D layout: VGPR i -> M = i + 8*lhi, N = ncol
        #pragma unroll
        for (int i = 0; i < 8; ++i) {
            int m = i + 8 * lhi;
            float* dst = Out + (size_t)outr[cur][m] * CCH + ncol;
            __hip_atomic_fetch_add(dst, acc[i], __ATOMIC_RELAXED,
                                   __HIP_MEMORY_SCOPE_AGENT);
        }
        __syncthreads();      // all reads of As[cur]/outr[cur] done before reuse
    }
}

// --- dense 1x1 conv: c2 = bf16(lrelu(lrelu(c1) @ W2)) -----------------------
// grid = N_PTS/(16*MTPB), block = 256
__global__ __launch_bounds__(256) void k_dense_mlp(
    const float*  __restrict__ Cin,    // c1 raw f32 [N,128]
    const __bf16* __restrict__ Wt2,    // [128(n)][128(kk)] bf16 transposed
    __bf16*       __restrict__ Cout)   // c2 bf16 [N,128]
{
    __shared__ __align__(16) __bf16 Bs[CCH * PITCH];
    __shared__ __align__(16) __bf16 As[16 * PITCH];

    const int t    = threadIdx.x;
    const int lane = t & 31;
    const int lhi  = lane >> 4;
    const int lm   = lane & 15;
    const int wv   = t >> 5;

    #pragma unroll
    for (int p = 0; p < 8; ++p) {
        int chk = p * 256 + t;
        __bf16*       dst = &Bs[(chk >> 4) * PITCH + (chk & 15) * 8];
        const __bf16* src = Wt2 + chk * 8;
#if HAVE_ASYNC_LDS
        async_copy16(src, dst);
#else
        *(uint4*)dst = *(const uint4*)src;
#endif
    }
#if HAVE_ASYNC_LDS
    wait_async0();
#endif
    __syncthreads();

    const int ncol = wv * 16 + lm;
    v16bf bfrag[4];
    #pragma unroll
    for (int s = 0; s < 4; ++s) {
        const __bf16* bp = &Bs[ncol * PITCH + s * 32 + lhi * 16];
        v8bf lo = *(const v8bf*)(bp);
        v8bf hi = *(const v8bf*)(bp + 8);
        #pragma unroll
        for (int j = 0; j < 8; ++j) { bfrag[s][j] = lo[j]; bfrag[s][j + 8] = hi[j]; }
    }

    for (int it = 0; it < MTPB; ++it) {
        const int m_base = (blockIdx.x * MTPB + it) * 16;
        __syncthreads();   // WAR: previous tile's As reads done
        // stage 16 rows, applying lrelu and converting f32 -> bf16
        #pragma unroll
        for (int rep = 0; rep < 2; ++rep) {
            int idx = rep * 256 + t;          // 512 chunks of 4 floats
            int r   = idx >> 5;
            int ch  = idx & 31;
            float4 v = *(const float4*)(Cin + (size_t)(m_base + r) * CCH + ch * 4);
            v4bf o;
            o[0] = (__bf16)lrelu(v.x); o[1] = (__bf16)lrelu(v.y);
            o[2] = (__bf16)lrelu(v.z); o[3] = (__bf16)lrelu(v.w);
            *(v4bf*)(&As[r * PITCH + ch * 4]) = o;
        }
        __syncthreads();

        v8f acc = {};
        #pragma unroll
        for (int s = 0; s < 4; ++s) {
            const __bf16* ap = &As[lm * PITCH + s * 32 + lhi * 16];
            v8bf lo = *(const v8bf*)(ap);
            v8bf hi = *(const v8bf*)(ap + 8);
            v16bf afrag;
            #pragma unroll
            for (int j = 0; j < 8; ++j) { afrag[j] = lo[j]; afrag[j + 8] = hi[j]; }
            acc = __builtin_amdgcn_wmma_f32_16x16x32_bf16(
                      false, afrag, false, bfrag[s], (short)0, acc, false, false);
        }

        #pragma unroll
        for (int i = 0; i < 8; ++i) {
            int m = i + 8 * lhi;
            Cout[(size_t)(m_base + m) * CCH + ncol] = (__bf16)lrelu(acc[i]);
        }
    }
}

// ---------------------------------------------------------------------------

extern "C" void kernel_launch(void* const* d_in, const int* in_sizes, int n_in,
                              void* d_out, int out_size, void* d_ws, size_t ws_size,
                              hipStream_t stream) {
    const float* feats   = (const float*)d_in[0];
    const float* W1      = (const float*)d_in[1];
    const float* W2      = (const float*)d_in[2];
    const float* W3      = (const float*)d_in[3];
    const int*   in_map  = (const int*)d_in[4];
    const int*   out_map = (const int*)d_in[5];
    const float* mask    = (const float*)d_in[6];
    float* out = (float*)d_out;

    char* ws = (char*)d_ws;
    const size_t featsb_bytes = (size_t)N_PTS * CCH * 2;        // 51.2 MB
    const size_t c1_bytes     = (size_t)N_PTS * CCH * 4;        // 102.4 MB
    __bf16* featsb = (__bf16*)ws;
    float*  c1     = (float*)(ws + featsb_bytes);
    __bf16* wt1    = (__bf16*)(ws + featsb_bytes + c1_bytes);
    __bf16* wt3    = wt1 + (size_t)KTAPS * CCH * CCH;
    __bf16* wt2    = wt3 + (size_t)KTAPS * CCH * CCH;
    __bf16* c2b    = featsb;   // alias: feats_bf16 dead after conv1

    // zero the two f32 accumulators (capture-safe)
    (void)hipMemsetAsync(c1, 0, c1_bytes, stream);
    (void)hipMemsetAsync(out, 0, (size_t)out_size * sizeof(float), stream);

    const int total4 = N_PTS * CCH / 4;   // 6,400,000
    k_convert_feats<<<(total4 + 255) / 256, 256, 0, stream>>>(feats, featsb, total4);
    k_transpose_w<<<(KTAPS * CCH * CCH + 255) / 256, 256, 0, stream>>>(W1, wt1, KTAPS * CCH * CCH);
    k_transpose_w<<<(KTAPS * CCH * CCH + 255) / 256, 256, 0, stream>>>(W3, wt3, KTAPS * CCH * CCH);
    k_transpose_w<<<(CCH * CCH + 255) / 256, 256, 0, stream>>>(W2, wt2, CCH * CCH);

    dim3 gconv(M_PAIRS / (16 * MTPB), KTAPS);      // (1250, 27)
    k_sparse_conv<<<gconv, 256, 0, stream>>>(featsb, wt1, in_map, out_map, c1);
    k_dense_mlp<<<N_PTS / (16 * MTPB), 256, 0, stream>>>(c1, wt2, c2b);
    k_sparse_conv<<<gconv, 256, 0, stream>>>(c2b, wt3, in_map, out_map, out);
    k_mask<<<(total4 + 255) / 256, 256, 0, stream>>>(out, mask, total4);

    (void)in_sizes; (void)n_in; (void)ws_size;
}
